// DecoderConv_15324443312547
// MI455X (gfx1250) — matrix operand order repaired and checked by reading, not compiled, not run on hardware
//
#include <hip/hip_runtime.h>
#include <stdint.h>

// ---------------- problem constants ----------------
#define B_      8
#define CIN     256
#define COUT    256
#define L_      16384
#define KS      9
#define PADK    4
#define NTOT    (B_ * L_)        // 131072
#define RCHUNKS 72               // (KS*CIN)/32 = 2304/32
#define NTILE   128
#define MTILE   256              // full Cout per block
#define NTILES  (NTOT / NTILE)   // 1024
#define XCOLS   (NTILE + KS - 1) // 136
#define XPITCH  140              // padded f32 pitch for x slab in LDS
#define BPITCH  40               // bf16 pitch (80B rows -> conflict-free frag reads)

typedef __attribute__((ext_vector_type(16))) __bf16 v16bf;
typedef __attribute__((ext_vector_type(8)))  float  v8f;

union ABits { uint4 q[2]; v16bf v; };   // 32B per-lane WMMA operand

__device__ __forceinline__ uint16_t f2bf(float f) {
  uint32_t u = __float_as_uint(f);
  uint32_t r = u + 0x7FFFu + ((u >> 16) & 1u);   // round-to-nearest-even
  return (uint16_t)(r >> 16);
}

#if __has_builtin(__builtin_amdgcn_cvt_pk_bf16_f32)
__device__ __forceinline__ uint32_t pk_bf16(float a, float b) {
  auto pk = __builtin_amdgcn_cvt_pk_bf16_f32(a, b);   // v_cvt_pk_bf16_f32
  return __builtin_bit_cast(uint32_t, pk);
}
#else
__device__ __forceinline__ uint32_t pk_bf16(float a, float b) {
  return (uint32_t)f2bf(a) | ((uint32_t)f2bf(b) << 16);
}
#endif

// ---------------- kernel 1: Gaussian window weights ----------------
__global__ void wgt_kernel(const float* __restrict__ coords, float* __restrict__ wgt) {
  int gid = blockIdx.x * blockDim.x + threadIdx.x;   // one (b,l) per thread
  int b = gid >> 14;
  int l = gid & (L_ - 1);
  const float* cb = coords + (size_t)b * 3 * L_;
  float cx = cb[l], cy = cb[L_ + l], cz = cb[2 * L_ + l];
  #pragma unroll
  for (int k = 0; k < KS; ++k) {
    int j = l + k - PADK;
    float d2;
    if (j >= 0 && j < L_) {
      float dx = cb[j] - cx, dy = cb[L_ + j] - cy, dz = cb[2 * L_ + j] - cz;
      d2 = dx * dx + dy * dy + dz * dz;
    } else {
      d2 = cx * cx + cy * cy + cz * cz;     // zero-padded window coords
    }
    wgt[((size_t)b * KS + k) * L_ + l] = __expf(-0.5f * d2);
  }
}

// ---------------- kernel 2: pack conv_w into WMMA A-fragment order ----------------
// Layout: pA[((r_chunk*16 + m_frag)*32 + lane)*16 + e]  (bf16)
// -> a wave's 4 fragments for one r_chunk are contiguous (immediate offsets).
__global__ void packA_kernel(const float* __restrict__ cw, uint16_t* __restrict__ pA) {
  int vidx    = blockIdx.x * blockDim.x + threadIdx.x;
  int i       = vidx & 15;
  int lane    = (vidx >> 4) & 31;
  int frag    = vidx >> 9;          // 0..1151
  int m_frag  = frag & 15;
  int r_chunk = frag >> 4;
  int row = m_frag * 16 + (lane & 15);
  // A 16x32 layout (contiguous k, matches B tile reads): lanes<16: K 0-15 ; lanes>=16: K 16-31
  int kk  = i + ((lane >> 4) << 4);
  int r   = r_chunk * 32 + kk;      // r = k*256 + c
  int kk9 = r >> 8;                 // kernel tap
  int c   = r & 255;                // cin
  pA[vidx] = f2bf(cw[((size_t)row * CIN + c) * KS + kk9]);
}

// ---------------- kernel 3: WMMA GEMM + bias + partial BN stats ----------------
__global__ __launch_bounds__(256)
void gemm_kernel(const float* __restrict__ x,
                 const uint16_t* __restrict__ pA,
                 const float* __restrict__ wgt,
                 const float* __restrict__ conv_b,
                 float* __restrict__ y,
                 float* __restrict__ psum,
                 float* __restrict__ psq) {
  __shared__ float    xs[32 * XPITCH];          // f32 x slab (17.9 KB)
  __shared__ float    wls[KS * NTILE];          // Gaussian weights tile
  __shared__ uint16_t bt[2][NTILE * BPITCH];    // double-buffered bf16 B tile
  __shared__ float    bias_s[MTILE];
  __shared__ float    red_s[MTILE];
  __shared__ float    red_q[MTILE];

  const int tid    = threadIdx.x;
  const int lane   = tid & 31;
  const int wave   = tid >> 5;
  const int wave_m = wave & 3;    // 4 wave rows x 64 channels
  const int wave_n = wave >> 2;   // 2 wave cols x 64 positions

  const int n_tile = blockIdx.x;
  const int b      = (n_tile * NTILE) >> 14;
  const int l0     = (n_tile * NTILE) & (L_ - 1);

  for (int idx = tid; idx < KS * NTILE; idx += 256) {
    int k = idx >> 7, n = idx & 127;
    wls[idx] = wgt[((size_t)b * KS + k) * L_ + l0 + n];
  }
  if (tid < MTILE) {
    bias_s[tid] = conv_b[tid];
    red_s[tid] = 0.f;
    red_q[tid] = 0.f;
  }

  v8f acc[4][4];
  #pragma unroll
  for (int i = 0; i < 4; ++i)
    #pragma unroll
    for (int j = 0; j < 4; ++j) acc[i][j] = (v8f)(0.f);

  const float* xb = x + (size_t)b * CIN * L_;
  // per-lane A base: fragment (r_chunk*16 + wave_m*4 + i), 1024 B per fragment
  const char* aBase = (const char*)pA + (size_t)wave_m * 4096 + (size_t)lane * 32;

  // x-staging mapping (division-free): 8 threads per row
  const int xrow  = tid >> 3;           // 0..31
  const int xcol  = (tid & 7) * 16;     // 0,16,...,112  (interior cols 4..131)
  const int hcol  = (tid & 7) < 4 ? (tid & 7) : (128 + (tid & 7));  // halo col

  // builder mapping (fixed per thread)
  const int bn    = tid & 127;          // n row of B tile this thread builds
  const int bc16  = (tid >> 7) << 4;    // c sub-block: 0 or 16
  // B-fragment read address pieces
  const int fn    = wave_n * 64 + (lane & 15);
  const int fc16  = (lane >> 4) << 4;

  #pragma unroll 1
  for (int cc = 0; cc < 8; ++cc) {          // 8 chunks of 32 input channels
    const int c0 = cc * 32;
    {
      // interior: gl = l0 + xcol + 0..15, always in range, 16B aligned
      const float4* src = (const float4*)(xb + (size_t)(c0 + xrow) * L_ + l0 + xcol);
      float4 v0 = src[0], v1 = src[1], v2 = src[2], v3 = src[3];
      float4* dst = (float4*)&xs[xrow * XPITCH + 4 + xcol];
      dst[0] = v0; dst[1] = v1; dst[2] = v2; dst[3] = v3;
      // halo: cols 0..3 and 132..135 (zero-padded at sequence edges)
      int gl = l0 - PADK + hcol;
      float hv = 0.f;
      if (gl >= 0 && gl < L_) hv = xb[(size_t)(c0 + xrow) * L_ + gl];
      xs[xrow * XPITCH + hcol] = hv;
    }
    __syncthreads();                        // xs ready (also: prior bt[p] reads done)
    #pragma unroll 1
    for (int k = 0; k < KS; ++k) {
      const int p = (cc + k) & 1;           // double-buffer parity
      {                                     // build bf16 B tile for this tap
        float w = wls[(k << 7) + bn];
        uint32_t* dst = (uint32_t*)&bt[p][bn * BPITCH + bc16];
        #pragma unroll
        for (int t = 0; t < 8; ++t) {
          float a0 = xs[(bc16 + 2 * t)     * XPITCH + bn + k] * w;
          float a1 = xs[(bc16 + 2 * t + 1) * XPITCH + bn + k] * w;
          dst[t] = pk_bf16(a0, a1);
        }
      }
      __syncthreads();                      // bt[p] ready (single barrier / iter)

      const uint4* ap = (const uint4*)(aBase + ((size_t)(k * 8 + cc) << 14));
      ABits a[4];
      #pragma unroll
      for (int i = 0; i < 4; ++i) {         // contiguous frags -> one load clause
        a[i].q[0] = ap[i * 64];
        a[i].q[1] = ap[i * 64 + 1];
      }
      ABits bb[4];
      #pragma unroll
      for (int j = 0; j < 4; ++j) {
        const uint4* src = (const uint4*)&bt[p][(fn + j * 16) * BPITCH + fc16];
        bb[j].q[0] = src[0];
        bb[j].q[1] = src[1];
      }
      #pragma unroll
      for (int i = 0; i < 4; ++i)
        #pragma unroll
        for (int j = 0; j < 4; ++j)
          acc[i][j] = __builtin_amdgcn_wmma_f32_16x16x32_bf16(
              false, a[i].v, false, bb[j].v, (short)0, acc[i][j], false, false);
    }
    __syncthreads();                        // all bt/xs reads done before next xs load
  }

  // epilogue: bias, store pre-BN y, per-channel partial sum / sumsq
  const int hi    = lane >> 4;
  const int col16 = lane & 15;
  #pragma unroll
  for (int i = 0; i < 4; ++i) {
    #pragma unroll
    for (int v = 0; v < 8; ++v) {
      int row = wave_m * 64 + i * 16 + v + hi * 8;     // C/D VGPR layout
      float bias = bias_s[row];
      size_t base = ((size_t)b * COUT + row) * L_ + l0 + wave_n * 64 + col16;
      float s = 0.f, q = 0.f;
      #pragma unroll
      for (int j = 0; j < 4; ++j) {
        float yv = acc[i][j][v] + bias;
        y[base + j * 16] = yv;
        s += yv;
        q += yv * yv;
      }
      #pragma unroll
      for (int msk = 1; msk < 16; msk <<= 1) {         // reduce across 16 N-columns
        s += __shfl_xor(s, msk, 32);
        q += __shfl_xor(q, msk, 32);
      }
      if (col16 == 0) {
        atomicAdd(&red_s[row], s);
        atomicAdd(&red_q[row], q);
      }
    }
  }
  __syncthreads();
  if (tid < MTILE) {
    psum[(size_t)tid * NTILES + n_tile] = red_s[tid];
    psq [(size_t)tid * NTILES + n_tile] = red_q[tid];
  }
}

// ---------------- kernel 4: finalize BN statistics ----------------
__global__ void stats_kernel(const float* __restrict__ psum, const float* __restrict__ psq,
                             const float* __restrict__ gamma, const float* __restrict__ beta,
                             float* __restrict__ scale, float* __restrict__ shift) {
  __shared__ float ss[128], qq[128];
  int ch = blockIdx.x, t = threadIdx.x;
  float s = 0.f, q = 0.f;
  for (int i = t; i < NTILES; i += 128) {
    s += psum[(size_t)ch * NTILES + i];
    q += psq [(size_t)ch * NTILES + i];
  }
  ss[t] = s; qq[t] = q;
  __syncthreads();
  for (int off = 64; off > 0; off >>= 1) {
    if (t < off) { ss[t] += ss[t + off]; qq[t] += qq[t + off]; }
    __syncthreads();
  }
  if (t == 0) {
    const float inv = 1.0f / (float)NTOT;
    float mean = ss[0] * inv;
    float var  = qq[0] * inv - mean * mean;   // biased variance
    float sc   = gamma[ch] * rsqrtf(var + 1e-5f);
    scale[ch] = sc;
    shift[ch] = beta[ch] - mean * sc;
  }
}

// ---------------- kernel 5: BN apply + ReLU, in-place on d_out ----------------
__global__ void apply_kernel(float* __restrict__ y, const float* __restrict__ scale,
                             const float* __restrict__ shift) {
  size_t idx  = (size_t)blockIdx.x * blockDim.x + threadIdx.x;
  size_t base = idx * 4;
  int o = (int)((base >> 14) & 255);
  float sc = scale[o], sh = shift[o];
  float4 v = ((float4*)y)[idx];
  v.x = fmaxf(0.f, v.x * sc + sh);
  v.y = fmaxf(0.f, v.y * sc + sh);
  v.z = fmaxf(0.f, v.z * sc + sh);
  v.w = fmaxf(0.f, v.w * sc + sh);
  ((float4*)y)[idx] = v;
}

// ---------------- launcher ----------------
extern "C" void kernel_launch(void* const* d_in, const int* in_sizes, int n_in,
                              void* d_out, int out_size, void* d_ws, size_t ws_size,
                              hipStream_t stream) {
  (void)in_sizes; (void)n_in; (void)out_size; (void)ws_size;
  const float* x      = (const float*)d_in[0];
  const float* coords = (const float*)d_in[1];
  const float* conv_w = (const float*)d_in[2];
  const float* conv_b = (const float*)d_in[3];
  const float* gamma  = (const float*)d_in[4];
  const float* beta   = (const float*)d_in[5];
  float* out = (float*)d_out;

  char* ws = (char*)d_ws;                       // ~8 MB of scratch used
  float*    wgt   = (float*)   (ws + 0);        // 4,718,592 B : Gaussian weights
  uint16_t* pA    = (uint16_t*)(ws + 4718592);  // 1,179,648 B : packed bf16 A
  float*    psum  = (float*)   (ws + 5898240);  // 1,048,576 B : partial sums
  float*    psq   = (float*)   (ws + 6946816);  // 1,048,576 B : partial sumsq
  float*    scale = (float*)   (ws + 7995392);  // 1,024 B
  float*    shift = (float*)   (ws + 7996416);  // 1,024 B

  wgt_kernel  <<<NTOT / 256, 256, 0, stream>>>(coords, wgt);
  packA_kernel<<<(16 * RCHUNKS * 512) / 256, 256, 0, stream>>>(conv_w, pA);
  gemm_kernel <<<NTILES, 256, 0, stream>>>(x, pA, wgt, conv_b, out, psum, psq);
  stats_kernel<<<COUT, 128, 0, stream>>>(psum, psq, gamma, beta, scale, shift);
  apply_kernel<<<(NTOT * COUT) / (256 * 4), 256, 0, stream>>>(out, scale, shift);
}
